// HierarchicalEEGTransformer_22960895164689
// MI455X (gfx1250) — matrix operand order repaired and checked by reading, not compiled, not run on hardware
//
#include <hip/hip_runtime.h>
#include <hip/hip_bf16.h>

typedef _Float16 v16h __attribute__((ext_vector_type(16)));
typedef _Float16 v8h  __attribute__((ext_vector_type(8)));
typedef float    v8f  __attribute__((ext_vector_type(8)));
typedef unsigned uint32x4 __attribute__((ext_vector_type(4)));
typedef int      int32x4  __attribute__((ext_vector_type(4)));
typedef int      int32x8  __attribute__((ext_vector_type(8)));

union AFrag { v16h v; v8h h[2]; };

__device__ __forceinline__ v8f wmma_f16(v16h a, v16h b, v8f c) {
  // D = A(16x32 f16) * B(32x16 f16) + C(16x16 f32)
  return __builtin_amdgcn_wmma_f32_16x16x32_f16(false, a, false, b, (short)0, c,
                                                false, false);
}
__device__ __forceinline__ v8f zero8() { v8f z = {}; return z; }

// ------------------------------------------------------------- TDM (CDNA5)
// Issue a 2D tensor-tile DMA: global (row-major, f16) -> LDS, tracked with
// TENSORcnt.  D# layout per CDNA5 ISA:
//   group0: count=1 | lds_addr | global_addr[56:0] | type=2
//   group1: data_size=1(2B), tensor_dim0=tile_w, tensor_dim1=tile_h,
//           tile_dim0=tile_w, tile_dim1=tile_h, tensor_dim0_stride=row_stride
__device__ __forceinline__ void tdm_load_2d_f16(const _Float16* gptr,
                                                unsigned lds_off,
                                                int tile_w, int tile_h,
                                                int row_stride) {
  unsigned long long ga = (unsigned long long)(size_t)gptr;
  uint32x4 g0;
  g0[0] = 1u;                                             // count=1 (user D#)
  g0[1] = lds_off;                                        // LDS byte address
  g0[2] = (unsigned)(ga & 0xffffffffull);                 // global_addr lo
  g0[3] = (unsigned)((ga >> 32) & 0x1ffffffull) | (2u << 30);  // hi | type=2
  int32x8 g1;
  g1[0] = 0x00010000;          // workgroup_mask=0, data_size=1 (2 bytes)
  g1[1] = tile_w << 16;        // tensor_dim0[15:0] at bits[63:48]
  g1[2] = tile_h << 16;        // tensor_dim1[15:0] at bits[95:80]
  g1[3] = tile_w << 16;        // tile_dim0 at bits[127:112]
  g1[4] = tile_h;              // tile_dim1 at bits[143:128]
  g1[5] = row_stride;          // tensor_dim0_stride[31:0]
  g1[6] = 0;
  g1[7] = 0;
  int32x4 z4 = {0, 0, 0, 0};
#if __clang_major__ >= 23
  int32x8 z8 = {0, 0, 0, 0, 0, 0, 0, 0};
  __builtin_amdgcn_tensor_load_to_lds(g0, g1, z4, z4, z8, 0);
#else
  __builtin_amdgcn_tensor_load_to_lds(g0, g1, z4, z4, 0);
#endif
}

// ---------------------------------------------------------------- utilities

__global__ __launch_bounds__(256)
void cast_transpose_kernel(const float* __restrict__ W, _Float16* __restrict__ Wt,
                           int K, int N) {
  // W: [K,N] f32 -> Wt: [N,K] f16
  long id = (long)blockIdx.x * 256 + threadIdx.x;
  if (id >= (long)K * N) return;
  int n = (int)(id / K), k = (int)(id % K);
  Wt[id] = (_Float16)W[(size_t)k * N + n];
}

__global__ __launch_bounds__(256)
void embed_kernel(const int* __restrict__ x, const float* __restrict__ emb,
                  const float* __restrict__ wpos, const float* __restrict__ tpos,
                  float* __restrict__ h) {
  // h[B*S, D] = emb[tok] + wpos[win] + tpos[pos];   S=1536, D=768, W=256
  int id = blockIdx.x * 256 + threadIdx.x;     // < 3072*768
  int d = id % 768;
  int r = id / 768;
  int b = r / 1536;
  int s = r % 1536;
  int win = s >> 8;
  int pos = s & 255;
  int tok = x[b * (6 * 257) + win * 257 + pos];
  h[id] = emb[(size_t)tok * 768 + d] + wpos[win * 768 + d] + tpos[pos * 768 + d];
}

__global__ __launch_bounds__(256)
void layernorm_kernel(const float* __restrict__ x, const float* __restrict__ g,
                      const float* __restrict__ bb, _Float16* __restrict__ y) {
  // one block per row of 768; output f16
  __shared__ float red[256];
  const int t = threadIdx.x;
  const size_t base = (size_t)blockIdx.x * 768;
  float a0 = x[base + t], a1 = x[base + t + 256], a2 = x[base + t + 512];
  red[t] = a0 + a1 + a2;
  __syncthreads();
  for (int o = 128; o > 0; o >>= 1) { if (t < o) red[t] += red[t + o]; __syncthreads(); }
  float mean = red[0] * (1.0f / 768.0f);
  __syncthreads();
  float d0 = a0 - mean, d1 = a1 - mean, d2 = a2 - mean;
  red[t] = d0 * d0 + d1 * d1 + d2 * d2;
  __syncthreads();
  for (int o = 128; o > 0; o >>= 1) { if (t < o) red[t] += red[t + o]; __syncthreads(); }
  float rstd = rsqrtf(red[0] * (1.0f / 768.0f) + 1e-5f);
  y[base + t]       = (_Float16)(d0 * rstd * g[t]       + bb[t]);
  y[base + t + 256] = (_Float16)(d1 * rstd * g[t + 256] + bb[t + 256]);
  y[base + t + 512] = (_Float16)(d2 * rstd * g[t + 512] + bb[t + 512]);
}

// ---------------------------------------------------------------- WMMA GEMM
// C[M,N] = A[M,K](f16, row-major) * Bt[N,K](f16, weight pre-transposed) + bias
// block = 256 threads = 8 waves; wave tile 32(M) x 64(N); block tile 64 x 256.
// A frag (contiguous k, same order as B): lane row = l&15, kbase = (l>=16)?16:0, halves
//   [kb+kbase .. +7] and [kb+kbase+8 .. +15]  -> two 16B loads.
// B frag: lane col = l&15, kbase = (l>=16)?16:0, halves [kb+kbase .. +15]
//   -> one 32B load from the transposed weight row.

enum { EPI_F32 = 0, EPI_F16 = 1, EPI_GELU = 2, EPI_RES = 3 };

template <int EPI>
__global__ __launch_bounds__(256)
void gemm_wmma(const _Float16* __restrict__ A, const _Float16* __restrict__ Bt,
               const float* __restrict__ bias, void* __restrict__ outp,
               float* __restrict__ resid, int M, int N, int K) {
  const int lane = threadIdx.x & 31;
  const int wave = threadIdx.x >> 5;
  const int wm = wave & 1, wn = wave >> 1;
  const int m0 = blockIdx.y * 64 + wm * 32;
  const int n0 = blockIdx.x * 256 + wn * 64;
  const int lr = lane & 15;
  const int hi = lane >> 4;

  v8f c[2][4];
#pragma unroll
  for (int mt = 0; mt < 2; ++mt)
#pragma unroll
    for (int nt = 0; nt < 4; ++nt) c[mt][nt] = zero8();

  for (int kb = 0; kb < K; kb += 32) {
    AFrag a[2];
#pragma unroll
    for (int mt = 0; mt < 2; ++mt) {
      const _Float16* ap = A + (size_t)(m0 + mt * 16 + lr) * K + kb + hi * 16;
      a[mt].h[0] = *(const v8h*)(ap);
      a[mt].h[1] = *(const v8h*)(ap + 8);
    }
    __builtin_prefetch(Bt + (size_t)(n0 + lr) * K + kb + 32, 0, 1);
#pragma unroll
    for (int nt = 0; nt < 4; ++nt) {
      const _Float16* bp = Bt + (size_t)(n0 + nt * 16 + lr) * K + kb + hi * 16;
      v16h b = *(const v16h*)(bp);
#pragma unroll
      for (int mt = 0; mt < 2; ++mt) c[mt][nt] = wmma_f16(a[mt].v, b, c[mt][nt]);
    }
  }

  // C/D layout: VGPR v -> row m0 + v + 8*hi, col = lane&15
#pragma unroll
  for (int mt = 0; mt < 2; ++mt)
#pragma unroll
    for (int nt = 0; nt < 4; ++nt)
#pragma unroll
      for (int vv = 0; vv < 8; ++vv) {
        int gr = m0 + mt * 16 + vv + hi * 8;
        int gc = n0 + nt * 16 + lr;
        float val = c[mt][nt][vv] + bias[gc];
        size_t o = (size_t)gr * N + gc;
        if (EPI == EPI_F32) {
          ((float*)outp)[o] = val;
        } else if (EPI == EPI_F16) {
          ((_Float16*)outp)[o] = (_Float16)val;
        } else if (EPI == EPI_GELU) {
          float gelu = 0.5f * val * (1.0f + erff(val * 0.70710678f));
          ((_Float16*)outp)[o] = (_Float16)gelu;
        } else {  // EPI_RES: residual add into f32 buffer
          resid[o] += val;
        }
      }
}

// ---------------------------------------------------------- flash attention
// grid = (qwin=6, head=12, b=2), block = 256 = 8 waves; wave owns 32 q-rows.
// Block-causal mask => window qwin attends exactly keys [0, (qwin+1)*256).
// K tile is staged global->LDS by the Tensor Data Mover (wave 0 issues the
// descriptor, waits on TENSORcnt); V tile is staged transposed by all threads
// in parallel with the DMA.

__global__ __launch_bounds__(256)
void attn_kernel(const _Float16* __restrict__ q, const _Float16* __restrict__ k,
                 const _Float16* __restrict__ v, _Float16* __restrict__ ctx) {
  __shared__ __align__(64) _Float16 kt_s[32 * 64];   // [key][d]   (TDM target)
  __shared__ __align__(64) _Float16 vt_s[64 * 32];   // [d][key]   (transposed)
  __shared__ __align__(64) _Float16 p_s[8][32 * 32]; // per-wave P tile [row][key]

  const int qwin = blockIdx.x;
  const int head = blockIdx.y;
  const int b    = blockIdx.z;
  const int lane = threadIdx.x & 31;
  const int wave = threadIdx.x >> 5;
  const int lr = lane & 15;
  const int hi = lane >> 4;
  const int hd0 = head * 64;
  const size_t rowbase = (size_t)b * 1536;
  const int qr0 = qwin * 256 + wave * 32;
  const unsigned kt_lds = (unsigned)(size_t)(&kt_s[0]);

  // Q fragments: [2 m-tiles][2 k-steps over HD=64]
  AFrag qa[2][2];
#pragma unroll
  for (int mt = 0; mt < 2; ++mt)
#pragma unroll
    for (int ks = 0; ks < 2; ++ks) {
      const _Float16* qp =
          q + (rowbase + qr0 + mt * 16 + lr) * 768 + hd0 + ks * 32 + hi * 16;
      qa[mt][ks].h[0] = *(const v8h*)(qp);
      qa[mt][ks].h[1] = *(const v8h*)(qp + 8);
    }

  v8f co[2][4];
#pragma unroll
  for (int mt = 0; mt < 2; ++mt)
#pragma unroll
    for (int nt = 0; nt < 4; ++nt) co[mt][nt] = zero8();
  float mi[2][8], li[2][8];
#pragma unroll
  for (int mt = 0; mt < 2; ++mt)
#pragma unroll
    for (int vv = 0; vv < 8; ++vv) { mi[mt][vv] = -1e30f; li[mt][vv] = 0.0f; }

  const int nk = (qwin + 1) * 256;
  for (int k0 = 0; k0 < nk; k0 += 32) {
    __syncthreads();  // previous tile fully consumed
    // K tile [32 x 64] via Tensor Data Mover (async, TENSORcnt-tracked)
    if (wave == 0) {
      tdm_load_2d_f16(k + (rowbase + k0) * 768 + hd0, kt_lds, 64, 32, 768);
    }
    // V tile staged transposed by all threads, overlapped with the DMA
    for (int i = threadIdx.x; i < 2048; i += 256) {
      int kk = i >> 6;
      int dd = i & 63;
      vt_s[dd * 32 + kk] = v[(rowbase + k0 + kk) * 768 + hd0 + dd];
    }
    if (wave == 0) __builtin_amdgcn_s_wait_tensorcnt(0);
    __syncthreads();

    // scores S[32q x 32k] = Q * K^T (K-dim = HD = 64 -> 2 chained wmma)
    v8f s[2][2];
#pragma unroll
    for (int t = 0; t < 2; ++t) {
      v16h kb0 = *(const v16h*)(kt_s + (t * 16 + lr) * 64 + 0 * 32 + hi * 16);
      v16h kb1 = *(const v16h*)(kt_s + (t * 16 + lr) * 64 + 1 * 32 + hi * 16);
#pragma unroll
      for (int mt = 0; mt < 2; ++mt) {
        v8f acc = zero8();
        acc = wmma_f16(qa[mt][0].v, kb0, acc);
        acc = wmma_f16(qa[mt][1].v, kb1, acc);
        s[mt][t] = acc;
      }
    }

    // online softmax (row stats live redundantly across the 16-lane group)
#pragma unroll
    for (int mt = 0; mt < 2; ++mt)
#pragma unroll
      for (int vv = 0; vv < 8; ++vv) {
        float x0 = s[mt][0][vv] * 0.125f;   // 1/sqrt(64)
        float x1 = s[mt][1][vv] * 0.125f;
        float mx = fmaxf(x0, x1);
        for (int o = 8; o >= 1; o >>= 1) mx = fmaxf(mx, __shfl_xor(mx, o));
        float mnew = fmaxf(mi[mt][vv], mx);
        float corr = __expf(mi[mt][vv] - mnew);
        float p0 = __expf(x0 - mnew);
        float p1 = __expf(x1 - mnew);
        float rs = p0 + p1;
        for (int o = 8; o >= 1; o >>= 1) rs += __shfl_xor(rs, o);
        li[mt][vv] = li[mt][vv] * corr + rs;
        mi[mt][vv] = mnew;
#pragma unroll
        for (int nt = 0; nt < 4; ++nt) co[mt][nt][vv] *= corr;
        int r = mt * 16 + vv + hi * 8;
        p_s[wave][r * 32 + lr]      = (_Float16)p0;
        p_s[wave][r * 32 + 16 + lr] = (_Float16)p1;
      }
    asm volatile("s_wait_dscnt 0" ::: "memory");  // wave-local P RAW in LDS

    // ctx += P[32x32] * V[32 x 64]
    const _Float16* ps = &p_s[wave][0];
    AFrag pa[2];
#pragma unroll
    for (int mt = 0; mt < 2; ++mt) {
      pa[mt].h[0] = *(const v8h*)(ps + (mt * 16 + lr) * 32 + hi * 16);
      pa[mt].h[1] = *(const v8h*)(ps + (mt * 16 + lr) * 32 + hi * 16 + 8);
    }
#pragma unroll
    for (int nt = 0; nt < 4; ++nt) {
      v16h vb = *(const v16h*)(vt_s + (nt * 16 + lr) * 32 + hi * 16);
#pragma unroll
      for (int mt = 0; mt < 2; ++mt) co[mt][nt] = wmma_f16(pa[mt].v, vb, co[mt][nt]);
    }
  }

  // normalize and store ctx (f16) into [B*S, D] at this head's slice
#pragma unroll
  for (int mt = 0; mt < 2; ++mt)
#pragma unroll
    for (int vv = 0; vv < 8; ++vv) {
      float inv = 1.0f / li[mt][vv];
      int gr = qr0 + mt * 16 + vv + hi * 8;
#pragma unroll
      for (int nt = 0; nt < 4; ++nt) {
        int gc = hd0 + nt * 16 + lr;
        ctx[(rowbase + gr) * 768 + gc] = (_Float16)(co[mt][nt][vv] * inv);
      }
    }
}

// ---------------------------------------------------------------- launcher

extern "C" void kernel_launch(void* const* d_in, const int* in_sizes, int n_in,
                              void* d_out, int out_size, void* d_ws, size_t ws_size,
                              hipStream_t stream) {
  (void)in_sizes; (void)n_in; (void)out_size; (void)ws_size;
  constexpr int D = 768, L = 2, V = 8192, M = 3072, F = 3072;

  const int*   x     = (const int*)  d_in[0];
  const float* emb   = (const float*)d_in[1];
  const float* wpos  = (const float*)d_in[2];
  const float* tpos  = (const float*)d_in[3];
  const float* qw    = (const float*)d_in[4];
  const float* qb    = (const float*)d_in[5];
  const float* kw    = (const float*)d_in[6];
  const float* kb    = (const float*)d_in[7];
  const float* vw    = (const float*)d_in[8];
  const float* vb    = (const float*)d_in[9];
  const float* ow    = (const float*)d_in[10];
  const float* ob    = (const float*)d_in[11];
  const float* ln1g  = (const float*)d_in[12];
  const float* ln1b  = (const float*)d_in[13];
  const float* ff1w  = (const float*)d_in[14];
  const float* ff1b  = (const float*)d_in[15];
  const float* ff2w  = (const float*)d_in[16];
  const float* ff2b  = (const float*)d_in[17];
  const float* ln2g  = (const float*)d_in[18];
  const float* ln2b  = (const float*)d_in[19];
  const float* lnfg  = (const float*)d_in[20];
  const float* lnfb  = (const float*)d_in[21];
  const float* headw = (const float*)d_in[22];
  const float* headb = (const float*)d_in[23];

  char* base = (char*)d_ws;
  size_t off = 0;
  auto alloc = [&](size_t bytes) -> void* {
    void* p = base + off;
    off = (off + bytes + 255) & ~(size_t)255;
    return p;
  };
  float*    h    = (float*)    alloc((size_t)M * D * 4);
  _Float16* y16  = (_Float16*) alloc((size_t)M * D * 2);
  _Float16* q16  = (_Float16*) alloc((size_t)M * D * 2);
  _Float16* k16  = (_Float16*) alloc((size_t)M * D * 2);
  _Float16* v16  = (_Float16*) alloc((size_t)M * D * 2);
  _Float16* c16  = (_Float16*) alloc((size_t)M * D * 2);
  _Float16* f16a = (_Float16*) alloc((size_t)M * F * 2);
  _Float16* wqT  = (_Float16*) alloc((size_t)L * D * D * 2);
  _Float16* wkT  = (_Float16*) alloc((size_t)L * D * D * 2);
  _Float16* wvT  = (_Float16*) alloc((size_t)L * D * D * 2);
  _Float16* woT  = (_Float16*) alloc((size_t)L * D * D * 2);
  _Float16* wf1T = (_Float16*) alloc((size_t)L * D * F * 2);
  _Float16* wf2T = (_Float16*) alloc((size_t)L * D * F * 2);
  _Float16* whT  = (_Float16*) alloc((size_t)D * V * 2);

  // --- weight precast + transpose (f32 [K,N] -> f16 [N,K])
  for (int i = 0; i < L; ++i) {
    size_t dd = (size_t)i * D * D, df = (size_t)i * D * F;
    cast_transpose_kernel<<<(D * D + 255) / 256, 256, 0, stream>>>(qw + dd, wqT + dd, D, D);
    cast_transpose_kernel<<<(D * D + 255) / 256, 256, 0, stream>>>(kw + dd, wkT + dd, D, D);
    cast_transpose_kernel<<<(D * D + 255) / 256, 256, 0, stream>>>(vw + dd, wvT + dd, D, D);
    cast_transpose_kernel<<<(D * D + 255) / 256, 256, 0, stream>>>(ow + dd, woT + dd, D, D);
    cast_transpose_kernel<<<(D * F + 255) / 256, 256, 0, stream>>>(ff1w + df, wf1T + df, D, F);
    cast_transpose_kernel<<<(D * F + 255) / 256, 256, 0, stream>>>(ff2w + df, wf2T + df, F, D);
  }
  cast_transpose_kernel<<<(D * V + 255) / 256, 256, 0, stream>>>(headw, whT, D, V);

  // --- embedding + positions
  embed_kernel<<<(M * D) / 256, 256, 0, stream>>>(x, emb, wpos, tpos, h);

  // --- transformer layers
  dim3 gD(D / 256, M / 64);  // N=768 GEMMs
  for (int i = 0; i < L; ++i) {
    size_t dd = (size_t)i * D * D, df = (size_t)i * D * F;
    layernorm_kernel<<<M, 256, 0, stream>>>(h, ln1g + i * D, ln1b + i * D, y16);
    gemm_wmma<EPI_F16><<<gD, 256, 0, stream>>>(y16, wqT + dd, qb + i * D, q16, nullptr, M, D, D);
    gemm_wmma<EPI_F16><<<gD, 256, 0, stream>>>(y16, wkT + dd, kb + i * D, k16, nullptr, M, D, D);
    gemm_wmma<EPI_F16><<<gD, 256, 0, stream>>>(y16, wvT + dd, vb + i * D, v16, nullptr, M, D, D);
    attn_kernel<<<dim3(6, 12, 2), 256, 0, stream>>>(q16, k16, v16, c16);
    gemm_wmma<EPI_RES><<<gD, 256, 0, stream>>>(c16, woT + dd, ob + i * D, nullptr, h, M, D, D);
    layernorm_kernel<<<M, 256, 0, stream>>>(h, ln2g + i * D, ln2b + i * D, y16);
    gemm_wmma<EPI_GELU><<<dim3(F / 256, M / 64), 256, 0, stream>>>(
        y16, wf1T + df, ff1b + (size_t)i * F, f16a, nullptr, M, F, D);
    gemm_wmma<EPI_RES><<<gD, 256, 0, stream>>>(f16a, wf2T + df, ff2b + i * D, nullptr, h, M, D, F);
  }

  // --- final LN + vocab head
  layernorm_kernel<<<M, 256, 0, stream>>>(h, lnfg, lnfb, y16);
  gemm_wmma<EPI_F32><<<dim3(V / 256, M / 64), 256, 0, stream>>>(
      y16, whT, headb, d_out, nullptr, M, V, D);
}
